// LSTM_54872502173841
// MI455X (gfx1250) — matrix operand order repaired and checked by reading, not compiled, not run on hardware
//
#include <hip/hip_runtime.h>
#include <hip/hip_bf16.h>
#include <math.h>

#define SEQ_LEN    128
#define INPUT_DIM  256
#define HIDDEN_DIM 1024
#define NUM_CLASSES 10
#define BATCH      512
#define KTOT       (INPUT_DIM + HIDDEN_DIM)   // 1280
#define FOURH      (4 * HIDDEN_DIM)           // 4096
#define NKSTEPS    (KTOT / 32)                // 40
#define LDA        40                         // padded LDS row stride (elems)
#define LDB        40

typedef __bf16 v16bf __attribute__((ext_vector_type(16)));
typedef float  v8f   __attribute__((ext_vector_type(8)));

union FragBF {
    v16bf v;
    uint4 q[2];
};

// ---------------------------------------------------------------------------
// One-time conversion kernels
// ---------------------------------------------------------------------------
__global__ void cvt_w_kernel(const float* __restrict__ g_w,
                             const float* __restrict__ i_w,
                             const float* __restrict__ f_w,
                             const float* __restrict__ o_w,
                             __hip_bfloat16* __restrict__ W16) {
    int idx = blockIdx.x * blockDim.x + threadIdx.x;
    const int total = FOURH * KTOT;
    if (idx >= total) return;
    int gate = idx / (HIDDEN_DIM * KTOT);
    int rem  = idx % (HIDDEN_DIM * KTOT);
    const float* src = (gate == 0) ? g_w : (gate == 1) ? i_w : (gate == 2) ? f_w : o_w;
    W16[idx] = __float2bfloat16(src[rem]);
}

__global__ void cvt_e_kernel(const float* __restrict__ embed,
                             __hip_bfloat16* __restrict__ E16) {
    int idx = blockIdx.x * blockDim.x + threadIdx.x;
    if (idx >= SEQ_LEN * INPUT_DIM) return;
    E16[idx] = __float2bfloat16(embed[idx]);
}

__global__ void init_state_kernel(float* __restrict__ c,
                                  float* __restrict__ hf,
                                  __hip_bfloat16* __restrict__ H16) {
    int idx = blockIdx.x * blockDim.x + threadIdx.x;
    if (idx >= BATCH * HIDDEN_DIM) return;
    c[idx]  = 0.0f;
    hf[idx] = 0.0f;
    H16[idx] = __float2bfloat16(0.0f);
}

// ---------------------------------------------------------------------------
// Per-step fused gate GEMM:  Z = [x_t | h] @ W^T   (bias added in gate kernel)
// Block = 64(M) x 256(N), 8 waves in 2x4; per K-step (32):
//   async-DMA A-tile (64x32) + B-tile (256x32) into LDS (double buffered),
//   each wave does 8 v_wmma_f32_16x16x32_bf16 from LDS fragments.
// ---------------------------------------------------------------------------
__global__ __launch_bounds__(256) void lstm_gemm_kernel(
    const int* __restrict__ x, int t,
    const __hip_bfloat16* __restrict__ E16,
    const __hip_bfloat16* __restrict__ H16,
    const __hip_bfloat16* __restrict__ W16,
    float* __restrict__ Z)
{
    __shared__ __hip_bfloat16 shA[2][64 * LDA];    // 2 x 5120 B
    __shared__ __hip_bfloat16 shB[2][256 * LDB];   // 2 x 20480 B

    const int tid   = threadIdx.x;
    const int lane  = tid & 31;
    const int wave  = tid >> 5;
    const int wm    = wave >> 2;                   // 0..1
    const int wn    = wave & 3;                    // 0..3
    const int m0blk = blockIdx.y * 64;
    const int n0blk = blockIdx.x * 256;

    // ---- per-thread staging assignments (one A b128 + four B b128 per step)
    const int arow = tid >> 2;                     // 0..63
    const int acol = (tid & 3) * 8;                // 0,8,16,24
    const int tok  = x[(m0blk + arow) * SEQ_LEN + t];
    const __hip_bfloat16* aSrcE = E16 + (size_t)tok * INPUT_DIM + acol;
    const __hip_bfloat16* aSrcH = H16 + (size_t)(m0blk + arow) * HIDDEN_DIM + acol;
    const unsigned ldsA0      = (unsigned)(uintptr_t)&shA[0][arow * LDA + acol];
    const unsigned ldsAstride = (unsigned)sizeof(shA[0]);

    const __hip_bfloat16* bSrc[4];
    unsigned ldsB0[4];
#pragma unroll
    for (int r = 0; r < 4; ++r) {
        const int chunk = tid + r * 256;
        const int brow  = chunk >> 2;              // 0..255
        const int bcol  = (chunk & 3) * 8;
        bSrc[r]  = W16 + (size_t)(n0blk + brow) * KTOT + bcol;
        ldsB0[r] = (unsigned)(uintptr_t)&shB[0][brow * LDB + bcol];
    }
    const unsigned ldsBstride = (unsigned)sizeof(shB[0]);

    auto issue = [&](int kt, int buf) {
        const int k0 = kt * 32;
        const __hip_bfloat16* pa =
            (k0 < INPUT_DIM) ? (aSrcE + k0) : (aSrcH + (k0 - INPUT_DIM));
        {
            const unsigned la = ldsA0 + (unsigned)buf * ldsAstride;
            const unsigned long long ga = (unsigned long long)(uintptr_t)pa;
            asm volatile("global_load_async_to_lds_b128 %0, %1, off"
                         :: "v"(la), "v"(ga) : "memory");
        }
#pragma unroll
        for (int r = 0; r < 4; ++r) {
            const unsigned lb = ldsB0[r] + (unsigned)buf * ldsBstride;
            const unsigned long long gb =
                (unsigned long long)(uintptr_t)(bSrc[r] + k0);
            asm volatile("global_load_async_to_lds_b128 %0, %1, off"
                         :: "v"(lb), "v"(gb) : "memory");
        }
    };

    // ---- fragment read positions within LDS tiles
    const int lrow  = lane & 15;
    const int lhalf = lane >> 4;
    const int aoff  = lhalf * 8;                   // A K sub-offset
    const int boff  = lhalf * 8;                   // B K sub-offset

    v8f acc[2][4];
    const v8f vzero = {};
#pragma unroll
    for (int mi = 0; mi < 2; ++mi)
#pragma unroll
        for (int ni = 0; ni < 4; ++ni)
            acc[mi][ni] = vzero;

    issue(0, 0);

    for (int kt = 0; kt < NKSTEPS; ++kt) {
        const int buf = kt & 1;
        asm volatile("s_wait_asynccnt 0x0" ::: "memory");
        __syncthreads();
        if (kt + 1 < NKSTEPS) issue(kt + 1, buf ^ 1);

        const __hip_bfloat16* Ab = shA[buf];
        const __hip_bfloat16* Bb = shB[buf];
        FragBF a[2], b[4];
#pragma unroll
        for (int mi = 0; mi < 2; ++mi) {
            const __hip_bfloat16* p = Ab + (wm * 32 + mi * 16 + lrow) * LDA + aoff;
            a[mi].q[0] = *(const uint4*)(p);
            a[mi].q[1] = *(const uint4*)(p + 16);
        }
#pragma unroll
        for (int ni = 0; ni < 4; ++ni) {
            const __hip_bfloat16* p = Bb + (wn * 64 + ni * 16 + lrow) * LDB + boff;
            b[ni].q[0] = *(const uint4*)(p);
            b[ni].q[1] = *(const uint4*)(p + 16);
        }
#pragma unroll
        for (int mi = 0; mi < 2; ++mi)
#pragma unroll
            for (int ni = 0; ni < 4; ++ni)
                acc[mi][ni] = __builtin_amdgcn_wmma_f32_16x16x32_bf16(
                    false, a[mi].v, false, b[ni].v,
                    (short)0, acc[mi][ni], false, false);
    }

    // ---- store: C/D layout -> lane half selects M+8, vgpr index = M&7 ----
    const int m0 = m0blk + wm * 32;
    const int n0 = n0blk + wn * 64;
#pragma unroll
    for (int mi = 0; mi < 2; ++mi) {
#pragma unroll
        for (int ni = 0; ni < 4; ++ni) {
            const int col   = n0 + ni * 16 + lrow;
            const int mbase = m0 + mi * 16 + lhalf * 8;
#pragma unroll
            for (int v = 0; v < 8; ++v)
                Z[(size_t)(mbase + v) * FOURH + col] = acc[mi][ni][v];
        }
    }
}

// ---------------------------------------------------------------------------
// Per-step gate nonlinearities + state update
// ---------------------------------------------------------------------------
__global__ void lstm_gates_kernel(const float* __restrict__ Z,
                                  const float* __restrict__ g_b,
                                  const float* __restrict__ i_b,
                                  const float* __restrict__ f_b,
                                  const float* __restrict__ o_b,
                                  float* __restrict__ c,
                                  float* __restrict__ hf,
                                  __hip_bfloat16* __restrict__ H16) {
    int idx = blockIdx.x * blockDim.x + threadIdx.x;
    if (idx >= BATCH * HIDDEN_DIM) return;
    const int b = idx / HIDDEN_DIM;
    const int j = idx % HIDDEN_DIM;
    const float* zr = Z + (size_t)b * FOURH;

    const float g  = tanhf(zr[j] + g_b[j]);
    const float ig = 1.0f / (1.0f + __expf(-(zr[HIDDEN_DIM + j]     + i_b[j])));
    const float fg = 1.0f / (1.0f + __expf(-(zr[2 * HIDDEN_DIM + j] + f_b[j])));
    const float og = 1.0f / (1.0f + __expf(-(zr[3 * HIDDEN_DIM + j] + o_b[j])));

    const float cn = g * ig + c[idx] * fg;
    const float hn = tanhf(cn) * og;
    c[idx]  = cn;
    hf[idx] = hn;
    H16[idx] = __float2bfloat16(hn);
}

// ---------------------------------------------------------------------------
// Classifier head + log_softmax
// ---------------------------------------------------------------------------
__global__ void head_kernel(const float* __restrict__ hf,
                            const float* __restrict__ p_w,
                            const float* __restrict__ p_b,
                            float* __restrict__ logits) {
    int idx = blockIdx.x * blockDim.x + threadIdx.x;
    if (idx >= BATCH * NUM_CLASSES) return;
    const int b  = idx / NUM_CLASSES;
    const int cl = idx % NUM_CLASSES;
    const float* hr = hf  + (size_t)b  * HIDDEN_DIM;
    const float* wr = p_w + (size_t)cl * HIDDEN_DIM;
    float s = p_b[cl];
    for (int k = 0; k < HIDDEN_DIM; ++k) s += hr[k] * wr[k];
    logits[idx] = s;
}

__global__ void logsoftmax_kernel(const float* __restrict__ logits,
                                  float* __restrict__ out) {
    int b = blockIdx.x * blockDim.x + threadIdx.x;
    if (b >= BATCH) return;
    const float* r = logits + b * NUM_CLASSES;
    float m = r[0];
    for (int i = 1; i < NUM_CLASSES; ++i) m = fmaxf(m, r[i]);
    float s = 0.0f;
    for (int i = 0; i < NUM_CLASSES; ++i) s += __expf(r[i] - m);
    const float ls = logf(s);
    for (int i = 0; i < NUM_CLASSES; ++i) out[b * NUM_CLASSES + i] = r[i] - m - ls;
}

// ---------------------------------------------------------------------------
extern "C" void kernel_launch(void* const* d_in, const int* in_sizes, int n_in,
                              void* d_out, int out_size, void* d_ws, size_t ws_size,
                              hipStream_t stream) {
    (void)in_sizes; (void)n_in; (void)out_size; (void)ws_size;

    const int*   x     = (const int*)  d_in[0];
    const float* embed = (const float*)d_in[1];
    const float* g_w   = (const float*)d_in[2];
    const float* i_w   = (const float*)d_in[3];
    const float* f_w   = (const float*)d_in[4];
    const float* o_w   = (const float*)d_in[5];
    const float* g_b   = (const float*)d_in[6];
    const float* i_b   = (const float*)d_in[7];
    const float* f_b   = (const float*)d_in[8];
    const float* o_b   = (const float*)d_in[9];
    const float* p_w   = (const float*)d_in[10];
    const float* p_b   = (const float*)d_in[11];
    float* out = (float*)d_out;

    char* ws = (char*)d_ws;
    auto alloc = [&](size_t bytes) -> char* {
        char* p = ws;
        ws += (bytes + 255) & ~(size_t)255;
        return p;
    };
    __hip_bfloat16* W16 = (__hip_bfloat16*)alloc((size_t)FOURH * KTOT * 2);       // 10.5 MB
    __hip_bfloat16* E16 = (__hip_bfloat16*)alloc((size_t)SEQ_LEN * INPUT_DIM * 2);// 64 KB
    __hip_bfloat16* H16 = (__hip_bfloat16*)alloc((size_t)BATCH * HIDDEN_DIM * 2); // 1 MB
    float* Z      = (float*)alloc((size_t)BATCH * FOURH * 4);                     // 8 MB
    float* c      = (float*)alloc((size_t)BATCH * HIDDEN_DIM * 4);                // 2 MB
    float* hf     = (float*)alloc((size_t)BATCH * HIDDEN_DIM * 4);                // 2 MB
    float* logits = (float*)alloc((size_t)BATCH * NUM_CLASSES * 4);               // 20 KB

    // one-time conversions (re-done every call: deterministic, stateless)
    {
        const int n = FOURH * KTOT;
        cvt_w_kernel<<<(n + 255) / 256, 256, 0, stream>>>(g_w, i_w, f_w, o_w, W16);
    }
    {
        const int n = SEQ_LEN * INPUT_DIM;
        cvt_e_kernel<<<(n + 255) / 256, 256, 0, stream>>>(embed, E16);
    }
    {
        const int n = BATCH * HIDDEN_DIM;
        init_state_kernel<<<(n + 255) / 256, 256, 0, stream>>>(c, hf, H16);
    }

    // sequential recurrence
    dim3 gemm_grid(FOURH / 256, BATCH / 64);   // 16 x 8
    const int n_gate = BATCH * HIDDEN_DIM;
    for (int t = 0; t < SEQ_LEN; ++t) {
        lstm_gemm_kernel<<<gemm_grid, 256, 0, stream>>>(x, t, E16, H16, W16, Z);
        lstm_gates_kernel<<<(n_gate + 255) / 256, 256, 0, stream>>>(
            Z, g_b, i_b, f_b, o_b, c, hf, H16);
    }

    // classifier head + log_softmax
    {
        const int n = BATCH * NUM_CLASSES;
        head_kernel<<<(n + 255) / 256, 256, 0, stream>>>(hf, p_w, p_b, logits);
        logsoftmax_kernel<<<(BATCH + 255) / 256, 256, 0, stream>>>(logits, out);
    }
}
